// GATNet_17970143166801
// MI455X (gfx1250) — matrix operand order repaired and checked by reading, not compiled, not run on hardware
//
#include <hip/hip_runtime.h>
#include <math.h>

#define NEG_SLOPE 0.2f
#define BN_EPS 1e-5f

typedef float v2f  __attribute__((ext_vector_type(2)));
typedef float v8f  __attribute__((ext_vector_type(8)));
typedef _Float16 v16h __attribute__((ext_vector_type(16)));

// ---------------------------------------------------------------------------
// helpers
// ---------------------------------------------------------------------------
__device__ __forceinline__ void atomicMaxF(float* addr, float v) {
    // ordered float max via sign-split integer atomics (works with -inf init)
    if (v >= 0.0f)
        atomicMax((int*)addr, __float_as_int(v));
    else
        atomicMin((unsigned int*)addr, __float_as_uint(v));
}

__device__ __forceinline__ void edge_sd(const int* __restrict__ ei, int e, int Eraw,
                                        int& s, int& d) {
    if (e < Eraw) { s = ei[e]; d = ei[Eraw + e]; }
    else          { s = e - Eraw; d = s; }            // appended self loops
}

__global__ void fill_kernel(float* __restrict__ p, float v, int n) {
    int i = blockIdx.x * blockDim.x + threadIdx.x;
    if (i < n) p[i] = v;
}

// ---------------------------------------------------------------------------
// layer 1 linear: in=2 -> out=128 (K too small for matrix ops; scalar FMA)
// ---------------------------------------------------------------------------
__global__ void lin1_kernel(const float* __restrict__ x,
                            const float* __restrict__ Wl, const float* __restrict__ bl,
                            const float* __restrict__ Wr, const float* __restrict__ br,
                            float* __restrict__ xl, float* __restrict__ xr, int N) {
    int idx = blockIdx.x * blockDim.x + threadIdx.x;      // over N*128
    if (idx >= N * 128) return;
    int n = idx >> 7, o = idx & 127;
    float x0 = x[n * 2], x1 = x[n * 2 + 1];
    xl[idx] = x0 * Wl[o] + x1 * Wl[128 + o] + bl[o];
    xr[idx] = x0 * Wr[o] + x1 * Wr[128 + o] + br[o];
}

// ---------------------------------------------------------------------------
// layer 2 linear via WMMA: h1[N,128] x W[128,64] (+bias) -> xl2/xr2[N,64]
// one wave computes a 16x16 output tile; waves cover (mtile, 4 ntiles x {l,r})
// ---------------------------------------------------------------------------
__global__ void lin2_wmma_kernel(const float* __restrict__ h,
                                 const float* __restrict__ Wl, const float* __restrict__ bl,
                                 const float* __restrict__ Wr, const float* __restrict__ br,
                                 float* __restrict__ xl, float* __restrict__ xr, int N) {
#if defined(__gfx1250__)
    int gtid  = blockIdx.x * blockDim.x + threadIdx.x;
    int wave  = gtid >> 5;
    int lane  = threadIdx.x & 31;
    int which = wave & 7;                  // 0..3 -> Wl/xl tile n, 4..7 -> Wr/xr
    int mtile = wave >> 3;
    bool isR  = which >= 4;
    int ntile = which & 3;

    const float* W    = isR ? Wr : Wl;
    const float* bias = isR ? br : bl;
    float*       out  = isR ? xr : xl;

    int row = mtile * 16 + (lane & 15);    // A row for this lane (16x4 layout)
    int arow = row < N ? row : N - 1;      // clamp (no divergence before WMMA)
    const float* ap = h + (size_t)arow * 128;
    int col  = ntile * 16 + (lane & 15);   // B/D column for this lane
    int koff = (lane < 16) ? 0 : 2;        // lanes 16-31 hold K+2,K+3

    v8f acc = {0.f, 0.f, 0.f, 0.f, 0.f, 0.f, 0.f, 0.f};

#if __has_builtin(__builtin_amdgcn_wmma_f32_16x16x4_f32)
    for (int k = 0; k < 128; k += 4) {
        v2f a, b;
        a.x = ap[k + koff];
        a.y = ap[k + koff + 1];
        b.x = W[(size_t)(k + koff)     * 64 + col];
        b.y = W[(size_t)(k + koff + 1) * 64 + col];
        acc = __builtin_amdgcn_wmma_f32_16x16x4_f32(
                  false, a, false, b, (short)0, acc, false, false);
    }
#else
    // fallback: f16 inputs, f32 accumulate (codegen-confirmed builtin)
    int kA = ((lane >> 4) << 3);           // A: lanes0-15 K{0..7,16..23}, hi +8
    int kB = ((lane >> 4) << 3);           // B: lanes0-15 K{0..7,16..23}, hi +8
    for (int kb = 0; kb < 128; kb += 32) {
        v16h a, b;
#pragma unroll
        for (int i = 0; i < 16; ++i) {
            int k = kb + (i & 7) + ((i >> 3) << 4) + kA;
            a[i] = (_Float16)ap[k];
        }
#pragma unroll
        for (int j = 0; j < 16; ++j) {
            int k = kb + (j & 7) + ((j >> 3) << 4) + kB;
            b[j] = (_Float16)W[(size_t)k * 64 + col];
        }
        acc = __builtin_amdgcn_wmma_f32_16x16x32_f16(
                  false, a, false, b, (short)0, acc, false, false);
    }
#endif

    // D layout: lane holds column (lane&15); rows v + 8*(lane>=16)
    int rbase = mtile * 16 + ((lane >> 4) << 3);
    float bb = bias[col];
#pragma unroll
    for (int v = 0; v < 8; ++v) {
        int r = rbase + v;
        if (r < N) out[(size_t)r * 64 + col] = acc[v] + bb;
    }
#endif // __gfx1250__
}

// ---------------------------------------------------------------------------
// layer 3 linear: in=64 -> out=2 (scalar)
// ---------------------------------------------------------------------------
__global__ void lin3_kernel(const float* __restrict__ h,
                            const float* __restrict__ Wl, const float* __restrict__ bl,
                            const float* __restrict__ Wr, const float* __restrict__ br,
                            float* __restrict__ xl, float* __restrict__ xr, int N) {
    int n = blockIdx.x * blockDim.x + threadIdx.x;
    if (n >= N) return;
    float al0 = bl[0], al1 = bl[1], ar0 = br[0], ar1 = br[1];
    const float* hp = h + (size_t)n * 64;
#pragma unroll 8
    for (int k = 0; k < 64; ++k) {
        float v = hp[k];
        al0 += v * Wl[k * 2];  al1 += v * Wl[k * 2 + 1];
        ar0 += v * Wr[k * 2];  ar1 += v * Wr[k * 2 + 1];
    }
    xl[n * 2] = al0; xl[n * 2 + 1] = al1;
    xr[n * 2] = ar0; xr[n * 2 + 1] = ar1;
}

// ---------------------------------------------------------------------------
// edge pass 1: logits = <att, leaky_relu(xl[src]+xr[dst])>, segment max
// ---------------------------------------------------------------------------
template <int H, int C>
__global__ void edge_logits_kernel(const int* __restrict__ ei, int Eraw, int Etot,
                                   const float* __restrict__ xl,
                                   const float* __restrict__ xr,
                                   const float* __restrict__ att,
                                   float* __restrict__ logits,
                                   float* __restrict__ segmax) {
    int e = blockIdx.x * blockDim.x + threadIdx.x;
    if (e >= Etot) return;
    int s, d; edge_sd(ei, e, Eraw, s, d);
    const float* pl = xl + (size_t)s * H * C;
    const float* pr = xr + (size_t)d * H * C;
#pragma unroll
    for (int h = 0; h < H; ++h) {
        float acc = 0.f;
#pragma unroll
        for (int c = 0; c < C; ++c) {
            float z = pl[h * C + c] + pr[h * C + c];
            z = z > 0.f ? z : NEG_SLOPE * z;
            acc += att[h * C + c] * z;
        }
        logits[(size_t)e * H + h] = acc;
        atomicMaxF(segmax + (size_t)d * H + h, acc);
    }
}

// ---------------------------------------------------------------------------
// edge pass 2: p = exp(logit - max[dst]); segment sum
// ---------------------------------------------------------------------------
template <int H>
__global__ void edge_exp_kernel(const int* __restrict__ ei, int Eraw, int Etot,
                                float* __restrict__ logits,
                                const float* __restrict__ segmax,
                                float* __restrict__ segsum) {
    int e = blockIdx.x * blockDim.x + threadIdx.x;
    if (e >= Etot) return;
    int s, d; edge_sd(ei, e, Eraw, s, d);
#pragma unroll
    for (int h = 0; h < H; ++h) {
        float p = expf(logits[(size_t)e * H + h] - segmax[(size_t)d * H + h]);
        logits[(size_t)e * H + h] = p;
        atomicAdd(segsum + (size_t)d * H + h, p);
    }
}

// ---------------------------------------------------------------------------
// edge pass 3: alpha = p / sum[dst]; out[dst] += alpha * xl[src]
// ---------------------------------------------------------------------------
template <int H, int C>
__global__ void edge_aggr_kernel(const int* __restrict__ ei, int Eraw, int Etot,
                                 const float* __restrict__ logits,
                                 const float* __restrict__ segsum,
                                 const float* __restrict__ xl,
                                 float* __restrict__ out) {
    int e = blockIdx.x * blockDim.x + threadIdx.x;
    if (e >= Etot) return;
    int s, d; edge_sd(ei, e, Eraw, s, d);
    const float* pl = xl + (size_t)s * H * C;
    float* po = out + (size_t)d * H * C;
#pragma unroll
    for (int h = 0; h < H; ++h) {
        float alpha = logits[(size_t)e * H + h] / segsum[(size_t)d * H + h];
#pragma unroll
        for (int c = 0; c < C; ++c)
            atomicAdd(po + h * C + c, alpha * pl[h * C + c]);
    }
}

// ---------------------------------------------------------------------------
// node post-processing
// ---------------------------------------------------------------------------
__global__ void add_bias_kernel(float* __restrict__ h, const float* __restrict__ bias,
                                int total, int HC) {
    int i = blockIdx.x * blockDim.x + threadIdx.x;
    if (i < total) h[i] += bias[i % HC];
}

__global__ void bn_stats_kernel(const float* __restrict__ h, int N, int HC,
                                float* __restrict__ stats) {
    int ch = blockIdx.x;                      // one block per channel
    int tid = threadIdx.x;
    float s = 0.f, s2 = 0.f;
    for (int n = tid; n < N; n += blockDim.x) {
        float v = h[(size_t)n * HC + ch];
        s += v; s2 += v * v;
    }
    __shared__ float sh[256], sh2[256];
    sh[tid] = s; sh2[tid] = s2;
    __syncthreads();
    for (int st = 128; st > 0; st >>= 1) {
        if (tid < st) { sh[tid] += sh[tid + st]; sh2[tid] += sh2[tid + st]; }
        __syncthreads();
    }
    if (tid == 0) {
        float mean = sh[0] / (float)N;
        float var  = sh2[0] / (float)N - mean * mean;
        stats[ch] = mean;
        stats[HC + ch] = var;
    }
}

__global__ void bn_elu_kernel(float* __restrict__ h, int total, int HC,
                              const float* __restrict__ stats,
                              const float* __restrict__ g,
                              const float* __restrict__ be) {
    int i = blockIdx.x * blockDim.x + threadIdx.x;
    if (i >= total) return;
    int ch = i % HC;
    float v = (h[i] - stats[ch]) * rsqrtf(stats[HC + ch] + BN_EPS) * g[ch] + be[ch];
    h[i] = v > 0.f ? v : expm1f(v);
}

__global__ void final_kernel(const float* __restrict__ h3,
                             const float* __restrict__ bias3,
                             float* __restrict__ out, int N) {
    int i = blockIdx.x * blockDim.x + threadIdx.x;
    if (i < 2 * N) out[i] = h3[i] + bias3[i & 1];  // H=1 -> mean is identity
}

// ---------------------------------------------------------------------------
// launch
// ---------------------------------------------------------------------------
extern "C" void kernel_launch(void* const* d_in, const int* in_sizes, int n_in,
                              void* d_out, int out_size, void* d_ws, size_t ws_size,
                              hipStream_t stream) {
    const float* x     = (const float*)d_in[0];
    const int*   ei    = (const int*)  d_in[1];
    const float* W1l   = (const float*)d_in[2];   const float* b1l   = (const float*)d_in[3];
    const float* W1r   = (const float*)d_in[4];   const float* b1r   = (const float*)d_in[5];
    const float* a1    = (const float*)d_in[6];   const float* bias1 = (const float*)d_in[7];
    const float* g1    = (const float*)d_in[8];   const float* be1   = (const float*)d_in[9];
    const float* W2l   = (const float*)d_in[10];  const float* b2l   = (const float*)d_in[11];
    const float* W2r   = (const float*)d_in[12];  const float* b2r   = (const float*)d_in[13];
    const float* a2    = (const float*)d_in[14];  const float* bias2 = (const float*)d_in[15];
    const float* g2    = (const float*)d_in[16];  const float* be2   = (const float*)d_in[17];
    const float* W3l   = (const float*)d_in[18];  const float* b3l   = (const float*)d_in[19];
    const float* W3r   = (const float*)d_in[20];  const float* b3r   = (const float*)d_in[21];
    const float* a3    = (const float*)d_in[22];  const float* bias3 = (const float*)d_in[23];

    const int N  = in_sizes[0] / 2;
    const int E  = in_sizes[1] / 2;
    const int ET = E + N;                         // with self loops

    float* ws = (float*)d_ws;
    float* xl1 = ws;  ws += (size_t)N * 128;
    float* xr1 = ws;  ws += (size_t)N * 128;
    float* h1  = ws;  ws += (size_t)N * 128;
    float* xl2 = ws;  ws += (size_t)N * 64;
    float* xr2 = ws;  ws += (size_t)N * 64;
    float* h2  = ws;  ws += (size_t)N * 64;
    float* xl3 = ws;  ws += (size_t)N * 2;
    float* xr3 = ws;  ws += (size_t)N * 2;
    float* h3  = ws;  ws += (size_t)N * 2;
    float* logits = ws; ws += (size_t)ET * 8;
    float* segmax = ws; ws += (size_t)N * 8;
    float* segsum = ws; ws += (size_t)N * 8;
    float* stats  = ws; ws += 256;

    const int TB = 256;
    auto cdiv = [](int a, int b) { return (a + b - 1) / b; };
    const float NEG_INF = -INFINITY;

    // ---------------- Layer 1: H=8, C=16 ----------------
    lin1_kernel<<<cdiv(N * 128, TB), TB, 0, stream>>>(x, W1l, b1l, W1r, b1r, xl1, xr1, N);
    fill_kernel<<<cdiv(N * 8,   TB), TB, 0, stream>>>(segmax, NEG_INF, N * 8);
    fill_kernel<<<cdiv(N * 8,   TB), TB, 0, stream>>>(segsum, 0.f,     N * 8);
    fill_kernel<<<cdiv(N * 128, TB), TB, 0, stream>>>(h1,     0.f,     N * 128);
    edge_logits_kernel<8, 16><<<cdiv(ET, TB), TB, 0, stream>>>(ei, E, ET, xl1, xr1, a1, logits, segmax);
    edge_exp_kernel<8>      <<<cdiv(ET, TB), TB, 0, stream>>>(ei, E, ET, logits, segmax, segsum);
    edge_aggr_kernel<8, 16> <<<cdiv(ET, TB), TB, 0, stream>>>(ei, E, ET, logits, segsum, xl1, h1);
    add_bias_kernel<<<cdiv(N * 128, TB), TB, 0, stream>>>(h1, bias1, N * 128, 128);
    bn_stats_kernel<<<128, 256, 0, stream>>>(h1, N, 128, stats);
    bn_elu_kernel<<<cdiv(N * 128, TB), TB, 0, stream>>>(h1, N * 128, 128, stats, g1, be1);

    // ---------------- Layer 2: H=4, C=16 (WMMA projections) ----------------
    {
        int mtiles = cdiv(N, 16);
        int waves  = mtiles * 8;                   // 4 ntiles x {l,r}
        lin2_wmma_kernel<<<cdiv(waves * 32, 128), 128, 0, stream>>>(h1, W2l, b2l, W2r, b2r, xl2, xr2, N);
    }
    fill_kernel<<<cdiv(N * 4,  TB), TB, 0, stream>>>(segmax, NEG_INF, N * 4);
    fill_kernel<<<cdiv(N * 4,  TB), TB, 0, stream>>>(segsum, 0.f,     N * 4);
    fill_kernel<<<cdiv(N * 64, TB), TB, 0, stream>>>(h2,     0.f,     N * 64);
    edge_logits_kernel<4, 16><<<cdiv(ET, TB), TB, 0, stream>>>(ei, E, ET, xl2, xr2, a2, logits, segmax);
    edge_exp_kernel<4>      <<<cdiv(ET, TB), TB, 0, stream>>>(ei, E, ET, logits, segmax, segsum);
    edge_aggr_kernel<4, 16> <<<cdiv(ET, TB), TB, 0, stream>>>(ei, E, ET, logits, segsum, xl2, h2);
    add_bias_kernel<<<cdiv(N * 64, TB), TB, 0, stream>>>(h2, bias2, N * 64, 64);
    bn_stats_kernel<<<64, 256, 0, stream>>>(h2, N, 64, stats);
    bn_elu_kernel<<<cdiv(N * 64, TB), TB, 0, stream>>>(h2, N * 64, 64, stats, g2, be2);

    // ---------------- Layer 3: H=1, C=2, mean + bias ----------------
    lin3_kernel<<<cdiv(N, TB), TB, 0, stream>>>(h2, W3l, b3l, W3r, b3r, xl3, xr3, N);
    fill_kernel<<<cdiv(N,     TB), TB, 0, stream>>>(segmax, NEG_INF, N);
    fill_kernel<<<cdiv(N,     TB), TB, 0, stream>>>(segsum, 0.f,     N);
    fill_kernel<<<cdiv(N * 2, TB), TB, 0, stream>>>(h3,     0.f,     N * 2);
    edge_logits_kernel<1, 2><<<cdiv(ET, TB), TB, 0, stream>>>(ei, E, ET, xl3, xr3, a3, logits, segmax);
    edge_exp_kernel<1>     <<<cdiv(ET, TB), TB, 0, stream>>>(ei, E, ET, logits, segmax, segsum);
    edge_aggr_kernel<1, 2> <<<cdiv(ET, TB), TB, 0, stream>>>(ei, E, ET, logits, segsum, xl3, h3);
    final_kernel<<<cdiv(N * 2, TB), TB, 0, stream>>>(h3, bias3, (float*)d_out, N);
}
